// HierarchicalLNN_38671885533668
// MI455X (gfx1250) — matrix operand order repaired and checked by reading, not compiled, not run on hardware
//
#include <hip/hip_runtime.h>
#include <hip/hip_bf16.h>
#include <stdint.h>

#define NL      11
#define BSZ     64
#define TSTEPS  64
#define FDIM    900
#define HDIM    128
#define UDIM    256
#define NOUT    1024
#define KT_X    29          // ceil(900/32) -> K padded to 928
#define KT_P    4           // 128/32
#define KT_H    8           // 256/32
#define NTILES  64          // 1024/16 N-tiles
#define NTILE_X (NL*KT_X*NTILES)   // 20416 packed B tiles for Wx
#define NTILE_H (NL*KT_H*NTILES)   // 5632  packed B tiles for Wh
#define NTILE_P (NL*KT_P*NTILES)   // 2816  packed B tiles for Wp

typedef __attribute__((ext_vector_type(16))) __bf16 v16bf;
typedef __attribute__((ext_vector_type(8)))  float  v8f;
typedef int vi4 __attribute__((vector_size(16)));   // b128 payload type for async builtin

union Frag { uint32_t u[8]; uint4 q[2]; v16bf bf; };

// ---- CDNA5 async global->LDS path (ASYNCcnt), guarded so compile can't fail
#if defined(__has_builtin)
#  if __has_builtin(__builtin_amdgcn_global_load_async_to_lds_b128)
#    define HAS_ASYNC_LDS 1
#  endif
#endif
#ifndef HAS_ASYNC_LDS
#  define HAS_ASYNC_LDS 0
#endif

#if HAS_ASYNC_LDS
#  if defined(__has_builtin) && __has_builtin(__builtin_amdgcn_s_wait_asynccnt)
#    define WAIT_ASYNC(n) __builtin_amdgcn_s_wait_asynccnt(n)
#  else
#    define WAIT_ASYNC(n) asm volatile("s_wait_asynccnt %0" ::"i"(n) : "memory")
#  endif
#  define ASYNC_CP128(gp, lp)                                             \
     __builtin_amdgcn_global_load_async_to_lds_b128(                      \
         (__attribute__((address_space(1))) vi4*)(gp),                    \
         (__attribute__((address_space(3))) vi4*)(lp), 0, 0)
#else
#  define WAIT_ASYNC(n)
#endif

__device__ __forceinline__ uint32_t f2bf(float f) {   // round-to-nearest-even fp32 -> bf16
  uint32_t u = __float_as_uint(f);
  return (u + 0x7FFFu + ((u >> 16) & 1u)) >> 16;
}
__device__ __forceinline__ uint32_t pack2(float a, float b) {
  return f2bf(a) | (f2bf(b) << 16);
}
__device__ __forceinline__ v8f wmma_bf16(const Frag& a, const Frag& b, v8f c) {
  // v_wmma_f32_16x16x32_bf16: (neg_a, A, neg_b, B, c_mod, C, reuse_a, reuse_b)
  return __builtin_amdgcn_wmma_f32_16x16x32_bf16(false, a.bf, false, b.bf,
                                                 (short)0, c, false, false);
}
__device__ __forceinline__ float sigmoidf_(float x) { return 1.f / (1.f + __expf(-x)); }

// ---------------------------------------------------------------------------
// Kernel 1: repack W1|W2|Wa|Wb columns (N order) and row-split Wx/Wh/Wp into
// bf16 per-lane B-fragment tiles (32x16 per tile, 32 lanes x 32 bytes).
// B layout: element (k,n): lane = n + 16*((k>>3)&1); bf16 slot p = (k&7) + 8*(k>>4).
// ---------------------------------------------------------------------------
__global__ __launch_bounds__(256) void pack_weights_kernel(
    const float* __restrict__ W1, const float* __restrict__ W2,
    const float* __restrict__ Wa, const float* __restrict__ Wb,
    uint32_t* __restrict__ PWx, uint32_t* __restrict__ PWh,
    uint32_t* __restrict__ PWp)
{
  const int gid  = blockIdx.x * 256 + threadIdx.x;
  const int tile = gid >> 5, lane = gid & 31;
  int l, kt, nt, krow0, kmax;
  uint32_t* dst;
  if (tile < NTILE_X) {
    l = tile / (KT_X * NTILES); int r = tile % (KT_X * NTILES);
    kt = r / NTILES; nt = r % NTILES;
    krow0 = kt * 32; kmax = FDIM;                 // x rows, zero-pad 900..927
    dst = PWx + (size_t)tile * 256;
  } else if (tile < NTILE_X + NTILE_H) {
    int ti = tile - NTILE_X;
    l = ti / (KT_H * NTILES); int r = ti % (KT_H * NTILES);
    kt = r / NTILES; nt = r % NTILES;
    krow0 = FDIM + HDIM + kt * 32; kmax = FDIM + HDIM + UDIM;   // h rows
    dst = PWh + (size_t)ti * 256;
  } else {
    int ti = tile - (NTILE_X + NTILE_H);
    l = ti / (KT_P * NTILES); int r = ti % (KT_P * NTILES);
    kt = r / NTILES; nt = r % NTILES;
    krow0 = FDIM + kt * 32; kmax = FDIM + HDIM;                 // prev_h rows
    dst = PWp + (size_t)ti * 256;
  }
  const float* Wsel = (nt < 16) ? W1 : (nt < 32) ? W2 : (nt < 48) ? Wa : Wb;
  const int ncol = (nt & 15) * 16 + (lane & 15);
  const int kb   = krow0 + 8 * (lane >> 4);
  const float* src = Wsel + (size_t)l * 1284 * 256 + ncol;
  uint32_t pk[8];
#pragma unroll
  for (int e = 0; e < 8; ++e) {
    int k0 = kb + 2 * (e & 3) + 16 * (e >> 2);
    float f0 = (k0     < kmax) ? src[(size_t)k0 * 256]       : 0.f;
    float f1 = (k0 + 1 < kmax) ? src[(size_t)(k0 + 1) * 256] : 0.f;
    pk[e] = pack2(f0, f1);
  }
  uint4* dq = (uint4*)(dst + lane * 8);
  dq[0] = make_uint4(pk[0], pk[1], pk[2], pk[3]);
  dq[1] = make_uint4(pk[4], pk[5], pk[6], pk[7]);
}

// Kernel 2: concat biases -> biasc[l][1024] = [b1|b2|ba|bb]
__global__ __launch_bounds__(256) void pack_bias_kernel(
    const float* __restrict__ b1, const float* __restrict__ b2,
    const float* __restrict__ ba, const float* __restrict__ bbv,
    float* __restrict__ biasc)
{
  int gid = blockIdx.x * 256 + threadIdx.x;
  if (gid >= NL * NOUT) return;
  int l = gid >> 10, n = gid & 1023;
  int blk = n >> 8, c = n & 255;
  const float* s = (blk == 0) ? b1 : (blk == 1) ? b2 : (blk == 2) ? ba : bbv;
  biasc[gid] = s[l * 256 + c];
}

// ---------------------------------------------------------------------------
// Kernel 3: Gx[l][t][b][n] = x[l,b,t,:] @ Wx  (bf16 WMMA, f32 accum)
// Block 256 thr = 8 waves, 128x128 output tile.
// A: fp32 -> bf16 VALU conversion into LDS (A-fragment layout).
// B: 8 KB contiguous packed-fragment slab per K-step, double-buffered in LDS,
//    filled with GLOBAL_LOAD_ASYNC_TO_LDS_B128 one K-step ahead (ASYNCcnt).
// ---------------------------------------------------------------------------
__global__ __launch_bounds__(256) void gemm_x_kernel(
    const float* __restrict__ x, const uint32_t* __restrict__ PWx,
    float* __restrict__ Gx)
{
  __shared__ __align__(16) uint32_t As[8][32][8];   // 8 KB A stage (frag layout)
  __shared__ __align__(16) uint32_t Bs[2][2048];    // 2 x 8 KB B stage (8 tiles)
  const int l = blockIdx.z, nb = blockIdx.y, mb = blockIdx.x;
  const int tid = threadIdx.x, lane = tid & 31, wave = tid >> 5;
  const int wm = wave & 1, wn = wave >> 1;          // 2(M) x 4(N) wave grid
  v8f acc[4][2];
  const v8f vzero = {0.f, 0.f, 0.f, 0.f, 0.f, 0.f, 0.f, 0.f};
#pragma unroll
  for (int mi = 0; mi < 4; ++mi) { acc[mi][0] = vzero; acc[mi][1] = vzero; }

  const float*    xl = x   + (size_t)l * 4096 * FDIM;
  const uint32_t* bl = PWx + (size_t)l * KT_X * NTILES * 256;
  const int smt = wave;                              // staging M-tile
  const int sm  = mb * 128 + smt * 16 + (lane & 15); // staging row (= b*64+t)
  const float* srow = xl + (size_t)sm * FDIM;

  // stage the 8KB B slab for K-step kt into Bs[kt&1] (256 thr x 32 B)
  auto stage_b = [&](int kt) {
    const uint32_t* g = bl + ((size_t)kt * NTILES + nb * 8) * 256 + tid * 8;
    uint32_t*       d = &Bs[kt & 1][tid * 8];
#if HAS_ASYNC_LDS
    ASYNC_CP128(g, d);
    ASYNC_CP128(g + 4, d + 4);
#else
    const uint4* gq = (const uint4*)g;
    uint4*       dq = (uint4*)d;
    dq[0] = gq[0]; dq[1] = gq[1];
#endif
  };

  stage_b(0);
  for (int kt = 0; kt < KT_X; ++kt) {
    { // stage A tile: fp32 -> bf16, A-fragment layout, k>=900 zero pad
      const int kbase = kt * 32 + 8 * (lane >> 4);
      uint32_t pk[8];
#pragma unroll
      for (int g = 0; g < 2; ++g)
#pragma unroll
        for (int e = 0; e < 4; ++e) {
          const int k0 = kbase + g * 16 + 2 * e;
          float f0 = (k0     < FDIM) ? srow[k0]     : 0.f;
          float f1 = (k0 + 1 < FDIM) ? srow[k0 + 1] : 0.f;
          pk[g * 4 + e] = pack2(f0, f1);
        }
#pragma unroll
      for (int e = 0; e < 8; ++e) As[smt][lane][e] = pk[e];
    }
    if (kt + 1 < KT_X) {          // issue next slab while this one is consumed
      stage_b(kt + 1);
      WAIT_ASYNC(2);              // oldest pair (current buffer) complete
    } else {
      WAIT_ASYNC(0);
    }
    __syncthreads();              // publish As + Bs[kt&1]
    Frag bf0, bf1;
    {
      const uint32_t* bt = &Bs[kt & 1][(wn * 2) * 256 + lane * 8];
      const uint4* bp0 = (const uint4*)bt;
      bf0.q[0] = bp0[0]; bf0.q[1] = bp0[1];
      const uint4* bp1 = (const uint4*)(bt + 256);
      bf1.q[0] = bp1[0]; bf1.q[1] = bp1[1];
    }
#pragma unroll
    for (int mi = 0; mi < 4; ++mi) {
      Frag a;
      const uint4* ap = (const uint4*)&As[wm * 4 + mi][lane][0];
      a.q[0] = ap[0]; a.q[1] = ap[1];
      acc[mi][0] = wmma_bf16(a, bf0, acc[mi][0]);
      acc[mi][1] = wmma_bf16(a, bf1, acc[mi][1]);
    }
    __syncthreads();              // reads of As/Bs done before next overwrite
  }
  // store, remapping row (b*64+t) -> [l][t][b][n] so the scan reads contiguously
#pragma unroll
  for (int mi = 0; mi < 4; ++mi) {
    const int m0 = mb * 128 + (wm * 4 + mi) * 16 + 8 * (lane >> 4);
#pragma unroll
    for (int ni = 0; ni < 2; ++ni) {
      const int n = nb * 128 + (wn * 2 + ni) * 16 + (lane & 15);
#pragma unroll
      for (int r = 0; r < 8; ++r) {
        const int m = m0 + r, b = m >> 6, t = m & 63;
        Gx[(((size_t)l * TSTEPS + t) * BSZ + b) * NOUT + n] = acc[mi][ni][r];
      }
    }
  }
}

// ---------------------------------------------------------------------------
// Kernel 4: sequential scan. One 1024-thread workgroup (32 waves).
// h lives in LDS as a bf16 A-fragment image (4 M-tiles x 8 K-tiles).
// Wave w: j-tile jt=w&15 across all 4 gate blocks (gate math lane-local),
// M-half mh=w>>4. Per layer end: head + prev_h@Wp (WMMA, K-tiles 0..3).
// ---------------------------------------------------------------------------
__global__ __launch_bounds__(1024) void scan_kernel(
    const uint32_t* __restrict__ PWh, const uint32_t* __restrict__ PWp,
    const float* __restrict__ biasc, const float* __restrict__ Gx,
    float* __restrict__ gpb, float* __restrict__ preds,
    const float* __restrict__ head_w, const float* __restrict__ head_b)
{
  __shared__ __align__(16) uint32_t hA[4][KT_H][32][8];  // 32 KB h fragment image
  __shared__ float hid[BSZ][HDIM];                       // 32 KB fp32 h[:, :128]
  const int tid = threadIdx.x, lane = tid & 31, wave = tid >> 5;
  const int jt = wave & 15, mh = wave >> 4;
  const v8f vzero = {0.f, 0.f, 0.f, 0.f, 0.f, 0.f, 0.f, 0.f};

  for (int i = tid; i < BSZ * NOUT; i += 1024) gpb[i] = biasc[i & (NOUT - 1)]; // layer 0: prev_h=0
  {
    uint32_t* p = &hA[0][0][0][0];
    for (int i = tid; i < 4 * KT_H * 32 * 8; i += 1024) p[i] = 0u;             // h0 = 0
  }
  __syncthreads();

  for (int l = 0; l < NL; ++l) {
    for (int t = 0; t < TSTEPS; ++t) {
      v8f acc[2][4];
#pragma unroll
      for (int mi = 0; mi < 2; ++mi)
#pragma unroll
        for (int blk = 0; blk < 4; ++blk) acc[mi][blk] = vzero;
      if (t + 1 < TSTEPS) {   // cover next timestep's 256KB Gx slab (global_prefetch_b8)
        const float* gnext = Gx + ((size_t)l * TSTEPS + (t + 1)) * BSZ * NOUT;
        __builtin_prefetch(gnext + (size_t)tid * 64, 0, 1);
      }
      const uint32_t* bbase = PWh + (size_t)l * KT_H * NTILES * 256;
#pragma unroll
      for (int kt = 0; kt < KT_H; ++kt) {
        Frag a0, a1;
        {
          const uint4* p0 = (const uint4*)&hA[mh * 2 + 0][kt][lane][0];
          a0.q[0] = p0[0]; a0.q[1] = p0[1];
          const uint4* p1 = (const uint4*)&hA[mh * 2 + 1][kt][lane][0];
          a1.q[0] = p1[0]; a1.q[1] = p1[1];
        }
#pragma unroll
        for (int blk = 0; blk < 4; ++blk) {
          const int nt = jt + 16 * blk;
          Frag bf;
          const uint4* bp = (const uint4*)(bbase + ((size_t)kt * NTILES + nt) * 256 + lane * 8);
          bf.q[0] = bp[0]; bf.q[1] = bp[1];
          acc[0][blk] = wmma_bf16(a0, bf, acc[0][blk]);
          acc[1][blk] = wmma_bf16(a1, bf, acc[1][blk]);
        }
      }
      __syncthreads();   // all waves finished reading old h image
      const float* gxp = Gx + ((size_t)l * TSTEPS + t) * BSZ * NOUT;
      const int j = jt * 16 + (lane & 15);
#pragma unroll
      for (int mi = 0; mi < 2; ++mi) {
        const int mbase = (mh * 2 + mi) * 16 + 8 * (lane >> 4);
#pragma unroll
        for (int r = 0; r < 8; ++r) {
          const int m = mbase + r;                       // batch row
          const float* gr = gxp + (size_t)m * NOUT;
          const float* pr = gpb + (size_t)m * NOUT;
          float v0 = acc[mi][0][r] + gr[j]       + pr[j];
          float v1 = acc[mi][1][r] + gr[j + 256] + pr[j + 256];
          float va = acc[mi][2][r] + gr[j + 512] + pr[j + 512];
          float vb = acc[mi][3][r] + gr[j + 768] + pr[j + 768];
          float ff1 = tanhf(v0), ff2 = tanhf(v1);
          float ti  = sigmoidf_(va + vb);
          float hn  = ff1 * (1.f - ti) + ti * ff2;
          // write back into A-fragment image (element (m, k=j))
          const int kl = j & 31;
          const int lp = (m & 15) + 16 * ((kl >> 3) & 1);
          const int p  = (kl & 7) + 8 * (kl >> 4);
          unsigned short* hp = (unsigned short*)&hA[m >> 4][j >> 5][lp][p >> 1];
          hp[p & 1] = (unsigned short)f2bf(hn);
          if (t == TSTEPS - 1 && j < HDIM) hid[m][j] = hn;
        }
      }
      __syncthreads();
    } // t

    // per-layer head: raw = hid @ head_w[l] + head_b[l]; sigmoid on col 2
    if (tid < BSZ * 3) {
      const int b = tid / 3, c = tid % 3;
      float s = head_b[l * 3 + c];
      for (int k = 0; k < HDIM; ++k) s += hid[b][k] * head_w[(l * HDIM + k) * 3 + c];
      if (c == 2) s = sigmoidf_(s);
      preds[((size_t)l * BSZ + b) * 3 + c] = s;
    }

    if (l + 1 < NL) {
      // gpb = hid @ Wp[l+1] + bias[l+1]  (K-tiles 0..3 of the h image = hid)
      v8f gacc[4][2];
#pragma unroll
      for (int mi = 0; mi < 4; ++mi) { gacc[mi][0] = vzero; gacc[mi][1] = vzero; }
      const uint32_t* pb = PWp + (size_t)(l + 1) * KT_P * NTILES * 256;
#pragma unroll
      for (int kt = 0; kt < KT_P; ++kt) {
        Frag a[4];
#pragma unroll
        for (int mi = 0; mi < 4; ++mi) {
          const uint4* p = (const uint4*)&hA[mi][kt][lane][0];
          a[mi].q[0] = p[0]; a[mi].q[1] = p[1];
        }
#pragma unroll
        for (int ni = 0; ni < 2; ++ni) {
          const int nt = wave * 2 + ni;
          Frag bf;
          const uint4* bp = (const uint4*)(pb + ((size_t)kt * NTILES + nt) * 256 + lane * 8);
          bf.q[0] = bp[0]; bf.q[1] = bp[1];
#pragma unroll
          for (int mi = 0; mi < 4; ++mi) gacc[mi][ni] = wmma_bf16(a[mi], bf, gacc[mi][ni]);
        }
      }
      const float* bnext = biasc + (size_t)(l + 1) * NOUT;
#pragma unroll
      for (int mi = 0; mi < 4; ++mi) {
        const int m0 = mi * 16 + 8 * (lane >> 4);
#pragma unroll
        for (int ni = 0; ni < 2; ++ni) {
          const int n = (wave * 2 + ni) * 16 + (lane & 15);
#pragma unroll
          for (int r = 0; r < 8; ++r)
            gpb[(size_t)(m0 + r) * NOUT + n] = gacc[mi][ni][r] + bnext[n];
        }
      }
      __syncthreads();
      uint32_t* p = &hA[0][0][0][0];
      for (int i = tid; i < 4 * KT_H * 32 * 8; i += 1024) p[i] = 0u;  // h0=0 next layer
      __syncthreads();
    }
  }
}

// ---------------------------------------------------------------------------
// Kernel 5: fusion MLP + heads + softmax. Tiny; plain VALU. Writes all outputs.
// ---------------------------------------------------------------------------
__global__ __launch_bounds__(256) void fusion_kernel(
    const float* __restrict__ preds, const float* __restrict__ market,
    const float* __restrict__ fw1, const float* __restrict__ fb1,
    const float* __restrict__ fw2, const float* __restrict__ fb2,
    const float* __restrict__ fw3, const float* __restrict__ fb3,
    const float* __restrict__ fweights,
    const float* __restrict__ hb_w1, const float* __restrict__ hb_b1,
    const float* __restrict__ hb_w2, const float* __restrict__ hb_b2,
    const float* __restrict__ ht_w1, const float* __restrict__ ht_b1,
    const float* __restrict__ ht_w2, const float* __restrict__ ht_b2,
    const float* __restrict__ er_w,  const float* __restrict__ er_b,
    const float* __restrict__ ov_w,  const float* __restrict__ ov_b,
    float* __restrict__ out)
{
  __shared__ float smem[64 * 48 + 64 * 128 + 64 * 64];   // 60 KB
  float* fin = smem;                     // [64][48] (45 used)
  float* fh1 = smem + 64 * 48;           // [64][128]
  float* fh  = smem + 64 * 48 + 64 * 128;// [64][64]
  float* t1  = smem;                     // overlay after fin/fh1 dead
  float* t2  = smem + 64 * 32;
  const int tid = threadIdx.x;
  float* out_pred = out;
  float* out_alp  = out + 192;
  float* out_fw   = out + 192 + 2112;
  float* out_hb   = out_fw + 11;
  float* out_ht   = out_hb + 64;
  float* out_er   = out_ht + 64;
  float* out_ov   = out_er + 64;

  for (int i = tid; i < 64 * 33; i += 256) {
    int b = i / 33, c = i % 33, l = c / 3, jj = c % 3;
    float v = preds[((size_t)l * 64 + b) * 3 + jj];
    fin[b * 48 + c] = v;
    out_alp[b * 33 + c] = v;
  }
  for (int i = tid; i < 64 * 12; i += 256)
    fin[(i / 12) * 48 + 33 + (i % 12)] = market[i];
  __syncthreads();
  for (int i = tid; i < 64 * 128; i += 256) {
    int b = i >> 7, o = i & 127;
    float s = fb1[o];
    for (int k = 0; k < 45; ++k) s += fin[b * 48 + k] * fw1[k * 128 + o];
    fh1[i] = fmaxf(s, 0.f);
  }
  __syncthreads();
  for (int i = tid; i < 64 * 64; i += 256) {
    int b = i >> 6, o = i & 63;
    float s = fb2[o];
    for (int k = 0; k < 128; ++k) s += fh1[b * 128 + k] * fw2[k * 64 + o];
    fh[i] = fmaxf(s, 0.f);
  }
  __syncthreads();
  if (tid < 192) {
    int b = tid / 3, c = tid % 3;
    float s = fb3[c];
    for (int k = 0; k < 64; ++k) s += fh[b * 64 + k] * fw3[k * 3 + c];
    if (c == 2) s = sigmoidf_(s);
    out_pred[b * 3 + c] = s;
  }
  __syncthreads();                        // fin/fh1 dead -> overlay t1/t2
  for (int i = tid; i < 64 * 32; i += 256) {
    int b = i >> 5, o = i & 31;
    float s1 = hb_b1[o], s2 = ht_b1[o];
    for (int k = 0; k < 64; ++k) {
      float f = fh[b * 64 + k];
      s1 += f * hb_w1[k * 32 + o];
      s2 += f * ht_w1[k * 32 + o];
    }
    t1[i] = fmaxf(s1, 0.f);
    t2[i] = fmaxf(s2, 0.f);
  }
  __syncthreads();
  if (tid < 64) {
    int b = tid;
    float s1 = hb_b2[0], s2 = ht_b2[0], s3 = er_b[0], s4 = ov_b[0];
    for (int k = 0; k < 32; ++k) { s1 += t1[b * 32 + k] * hb_w2[k]; s2 += t2[b * 32 + k] * ht_w2[k]; }
    for (int k = 0; k < 64; ++k) { float f = fh[b * 64 + k]; s3 += f * er_w[k]; s4 += f * ov_w[k]; }
    out_hb[b] = sigmoidf_(s1);
    out_ht[b] = sigmoidf_(s2);
    out_er[b] = s3;
    out_ov[b] = s4;
  }
  if (tid == 0) {
    float mx = fweights[0];
    for (int i = 1; i < 11; ++i) mx = fmaxf(mx, fweights[i]);
    float e[11], sum = 0.f;
    for (int i = 0; i < 11; ++i) { e[i] = __expf(fweights[i] - mx); sum += e[i]; }
    for (int i = 0; i < 11; ++i) out_fw[i] = e[i] / sum;
  }
}

// ---------------------------------------------------------------------------
extern "C" void kernel_launch(void* const* d_in, const int* in_sizes, int n_in,
                              void* d_out, int out_size, void* d_ws, size_t ws_size,
                              hipStream_t stream)
{
  (void)in_sizes; (void)n_in; (void)out_size; (void)ws_size;
  const float* x      = (const float*)d_in[0];
  const float* market = (const float*)d_in[1];
  const float* W1 = (const float*)d_in[2];  const float* b1 = (const float*)d_in[3];
  const float* W2 = (const float*)d_in[4];  const float* b2 = (const float*)d_in[5];
  const float* Wa = (const float*)d_in[6];  const float* ba = (const float*)d_in[7];
  const float* Wb = (const float*)d_in[8];  const float* bb = (const float*)d_in[9];
  const float* head_w = (const float*)d_in[10];
  const float* head_b = (const float*)d_in[11];
  const float* fw1 = (const float*)d_in[12]; const float* fb1 = (const float*)d_in[13];
  const float* fw2 = (const float*)d_in[14]; const float* fb2 = (const float*)d_in[15];
  const float* fw3 = (const float*)d_in[16]; const float* fb3 = (const float*)d_in[17];
  const float* fweights = (const float*)d_in[18];
  const float* hb_w1 = (const float*)d_in[19]; const float* hb_b1 = (const float*)d_in[20];
  const float* hb_w2 = (const float*)d_in[21]; const float* hb_b2 = (const float*)d_in[22];
  const float* ht_w1 = (const float*)d_in[23]; const float* ht_b1 = (const float*)d_in[24];
  const float* ht_w2 = (const float*)d_in[25]; const float* ht_b2 = (const float*)d_in[26];
  const float* er_w = (const float*)d_in[27];  const float* er_b = (const float*)d_in[28];
  const float* ov_w = (const float*)d_in[29];  const float* ov_b = (const float*)d_in[30];

  char* ws = (char*)d_ws;
  uint32_t* PWx  = (uint32_t*)ws;
  uint32_t* PWh  = (uint32_t*)(ws + (size_t)NTILE_X * 1024);
  uint32_t* PWp  = (uint32_t*)(ws + (size_t)(NTILE_X + NTILE_H) * 1024);
  float*    biasc= (float*)(ws + (size_t)(NTILE_X + NTILE_H + NTILE_P) * 1024);
  float*    Gx   = (float*)((char*)biasc + (size_t)NL * NOUT * 4);
  float*    gpb  = (float*)((char*)Gx + (size_t)NL * TSTEPS * BSZ * NOUT * 4);
  float*    preds= (float*)((char*)gpb + (size_t)BSZ * NOUT * 4);

  const int packThreads = (NTILE_X + NTILE_H + NTILE_P) * 32;   // 923648
  pack_weights_kernel<<<packThreads / 256, 256, 0, stream>>>(W1, W2, Wa, Wb, PWx, PWh, PWp);
  pack_bias_kernel<<<(NL * NOUT + 255) / 256, 256, 0, stream>>>(b1, b2, ba, bb, biasc);

  dim3 gg(32, 8, NL);   // 4096 rows / 128, 1024 cols / 128, layers
  gemm_x_kernel<<<gg, 256, 0, stream>>>(x, PWx, Gx);

  scan_kernel<<<1, 1024, 0, stream>>>(PWh, PWp, biasc, Gx, gpb, preds, head_w, head_b);

  fusion_kernel<<<1, 256, 0, stream>>>(preds, market,
      fw1, fb1, fw2, fb2, fw3, fb3, fweights,
      hb_w1, hb_b1, hb_w2, hb_b2, ht_w1, ht_b1, ht_w2, ht_b2,
      er_w, er_b, ov_w, ov_b, (float*)d_out);
}
